// JointNet_64415919506129
// MI455X (gfx1250) — matrix operand order repaired and checked by reading, not compiled, not run on hardware
//
#include <hip/hip_runtime.h>
#include <hip/hip_fp16.h>
#include <math.h>

typedef __attribute__((ext_vector_type(16))) _Float16 v16h;
typedef __attribute__((ext_vector_type(8)))  _Float16 v8h;
typedef __attribute__((ext_vector_type(8)))  float    v8f;

#define BB   4
#define TT   150
#define UU   40
#define DI   512
#define VV   4000
#define M_TOTAL (BB*TT*UU)    /* 24000, divisible by 16 */
#define NTILES  (VV/16)       /* 250 */
#define KSTEPS  (DI/32)       /* 16  */

// ---------------- Kernel A: out[row,col] = dot(state[row,:], W[col,:]) + bias[col]
__global__ void prelinear_kernel(const float* __restrict__ st,
                                 const float* __restrict__ W,
                                 const float* __restrict__ bias,
                                 float* __restrict__ out, int rows) {
  int o = blockIdx.x * blockDim.x + threadIdx.x;
  if (o >= rows * DI) return;
  int row = o >> 9;
  int col = o & (DI - 1);
  const float4* a = (const float4*)(st + (size_t)row * DI);
  const float4* w = (const float4*)(W  + (size_t)col * DI);
  float acc = 0.f;
#pragma unroll 4
  for (int k = 0; k < DI / 4; ++k) {
    float4 av = a[k], wv = w[k];
    acc = fmaf(av.x, wv.x, acc);
    acc = fmaf(av.y, wv.y, acc);
    acc = fmaf(av.z, wv.z, acc);
    acc = fmaf(av.w, wv.w, acc);
  }
  out[o] = acc + bias[col];
}

// ---------------- Kernel B: W_proj fp32 -> f16 (row-major [V, DI])
__global__ void packw_kernel(const float* __restrict__ in, _Float16* __restrict__ outp, int n) {
  int i = blockIdx.x * blockDim.x + threadIdx.x;
  if (i < n) outp[i] = (_Float16)in[i];
}

// ---------------- Kernel C: fused joint-tanh + WMMA vocab GEMM + log-softmax
__global__ __launch_bounds__(256)
void joint_kernel(const float* __restrict__ encf,     // [B*T, DI] fp32
                  const float* __restrict__ decf,     // [B*U, DI] fp32
                  const _Float16* __restrict__ wp,    // [V, DI]  f16
                  const float* __restrict__ bproj,    // [V]
                  float* __restrict__ out)            // [M_TOTAL, V]
{
  extern __shared__ char smem[];
  _Float16* lds_joint  = (_Float16*)smem;                              // 16*512 f16 = 16 KB
  float*    lds_logits = (float*)(smem + 16 * DI * sizeof(_Float16));  // 16*4000 f32 = 250 KB

  const int tid  = threadIdx.x;
  const int lane = tid & 31;
  const int wave = tid >> 5;
  const int m0   = blockIdx.x * 16;

  // ---- build 16x512 f16 joint tile: tanh(enc[bt,:] + dec[b*U+u,:])
  {
    const int r  = tid >> 4;           // row 0..15
    const int k0 = (tid & 15) * 32;    // 32 contiguous k per thread
    const int m  = m0 + r;
    const int u  = m % UU;
    const int bt = m / UU;
    const int b  = m / (TT * UU);
    const float4* e4 = (const float4*)(encf + (size_t)bt * DI + k0);
    const float4* d4 = (const float4*)(decf + (size_t)(b * UU + u) * DI + k0);
    _Float16* jp = lds_joint + r * DI + k0;
#pragma unroll
    for (int q = 0; q < 8; ++q) {
      float4 ev = e4[q], dv = d4[q];
      jp[q * 4 + 0] = (_Float16)tanhf(ev.x + dv.x);
      jp[q * 4 + 1] = (_Float16)tanhf(ev.y + dv.y);
      jp[q * 4 + 2] = (_Float16)tanhf(ev.z + dv.z);
      jp[q * 4 + 3] = (_Float16)tanhf(ev.w + dv.w);
    }
  }
  __syncthreads();

  // ---- preload all 16 A fragments (ISA 16-bit 16x32 A layout)
  const int mlane  = lane & 15;
  const int kbaseA = (lane >> 4) << 3;   // 0 or 8
  const int kbaseB = (lane >> 4) << 3;   // 0 or 8
  v16h afrag[KSTEPS];
#pragma unroll
  for (int ks = 0; ks < KSTEPS; ++ks) {
    const _Float16* p = lds_joint + mlane * DI + ks * 32 + kbaseA;
    union { v16h v; v8h h[2]; } ua;
    ua.h[0] = *(const v8h*)p;          // K = ks*32 + kbaseA + 0..7
    ua.h[1] = *(const v8h*)(p + 16);   // K = ks*32 + 16 + kbaseA + 0..7
    afrag[ks] = ua.v;
  }

  // ---- GEMM: N-tiles round-robin over 8 waves, K chained in fp32 accumulator
  for (int nt = wave; nt < NTILES; nt += 8) {
    const int ncol = nt * 16 + mlane;
    const _Float16* wrow = wp + (size_t)ncol * DI + kbaseB;
    v8f c = {};
#pragma unroll
    for (int ks = 0; ks < KSTEPS; ++ks) {
      union { v16h v; v8h h[2]; } ub;      // B 32x16: lane half covers K kbaseB+0..7 / 16+kbaseB+0..7
      ub.h[0] = *(const v8h*)(wrow + ks * 32);
      ub.h[1] = *(const v8h*)(wrow + ks * 32 + 16);
      c = __builtin_amdgcn_wmma_f32_16x16x32_f16(false, afrag[ks], false, ub.v,
                                                 (short)0, c, false, false);
    }
    const float bb   = bproj[ncol];
    const int mbase  = (lane < 16) ? 0 : 8;
    float* lp = lds_logits + nt * 16 + mlane;
#pragma unroll
    for (int g = 0; g < 8; ++g)
      lp[(size_t)(mbase + g) * VV] = c[g] + bb;
  }
  __syncthreads();

  // ---- log-softmax per row (wave w owns rows 2w, 2w+1), write final output
#pragma unroll
  for (int rr = 0; rr < 2; ++rr) {
    const int r = wave * 2 + rr;
    const float* Lr = lds_logits + (size_t)r * VV;
    float mx = -INFINITY;
    for (int j = lane; j < VV; j += 32) mx = fmaxf(mx, Lr[j]);
#pragma unroll
    for (int off = 16; off > 0; off >>= 1) mx = fmaxf(mx, __shfl_xor(mx, off, 32));
    float s = 0.f;
    for (int j = lane; j < VV; j += 32) s += expf(Lr[j] - mx);
#pragma unroll
    for (int off = 16; off > 0; off >>= 1) s += __shfl_xor(s, off, 32);
    const float lse = mx + logf(s);
    float* orow = out + (size_t)(m0 + r) * VV;
    for (int j = lane; j < VV; j += 32) orow[j] = Lr[j] - lse;
  }
}

extern "C" void kernel_launch(void* const* d_in, const int* in_sizes, int n_in,
                              void* d_out, int out_size, void* d_ws, size_t ws_size,
                              hipStream_t stream) {
  const float* enc_state = (const float*)d_in[0];
  const float* dec_state = (const float*)d_in[1];
  const float* W_enc     = (const float*)d_in[2];
  const float* b_enc     = (const float*)d_in[3];
  const float* W_prd     = (const float*)d_in[4];
  const float* b_prd     = (const float*)d_in[5];
  const float* W_proj    = (const float*)d_in[6];
  const float* b_proj    = (const float*)d_in[7];
  float* out = (float*)d_out;

  char* ws = (char*)d_ws;
  float*    encf = (float*)ws;                                      // 600*512 fp32
  float*    decf = (float*)(ws + (size_t)600 * 512 * 4);            // 160*512 fp32
  _Float16* wp   = (_Float16*)(ws + (size_t)(600 + 160) * 512 * 4); // 4000*512 f16

  {
    int n = (BB * TT) * DI;   // 307200
    prelinear_kernel<<<(n + 255) / 256, 256, 0, stream>>>(enc_state, W_enc, b_enc, encf, BB * TT);
  }
  {
    int n = (BB * UU) * DI;   // 81920
    prelinear_kernel<<<(n + 255) / 256, 256, 0, stream>>>(dec_state, W_prd, b_prd, decf, BB * UU);
  }
  {
    int n = VV * DI;          // 2048000
    packw_kernel<<<(n + 255) / 256, 256, 0, stream>>>(W_proj, wp, n);
  }

  size_t lds_bytes = (size_t)16 * DI * sizeof(_Float16) + (size_t)16 * VV * sizeof(float); // 272384
  (void)hipFuncSetAttribute((const void*)joint_kernel,
                            hipFuncAttributeMaxDynamicSharedMemorySize, (int)lds_bytes);
  joint_kernel<<<M_TOTAL / 16, 256, lds_bytes, stream>>>(encf, decf, wp, b_proj, out);
}
